// Transformer_63419487092956
// MI455X (gfx1250) — matrix operand order repaired and checked by reading, not compiled, not run on hardware
//
#include <hip/hip_runtime.h>
#include <math.h>

// ---------------------------------------------------------------------------
// Types for CDNA5 WMMA (gfx1250, wave32)
// ---------------------------------------------------------------------------
typedef __attribute__((ext_vector_type(16))) __bf16 bvec16;
typedef __attribute__((ext_vector_type(8)))  __bf16 bvec8;
typedef __attribute__((ext_vector_type(2)))  __bf16 bvec2;
typedef __attribute__((ext_vector_type(8)))  float  v8f;

constexpr int DMC   = 512;
constexpr int DFFC  = 2048;
constexpr int NHC   = 8;
constexpr int LSEQ  = 512;   // LS == LT
constexpr int BATCH = 8;
constexpr int NBLK  = 6;
constexpr int VOCAB = 32000;
constexpr int BL    = BATCH * LSEQ; // 4096 rows

__device__ __forceinline__ v8f wmma_bf16(bvec16 a, bvec16 b, v8f c) {
  return __builtin_amdgcn_wmma_f32_16x16x32_bf16(false, a, false, b, (short)0, c,
                                                 false, false);
}

// fp32 -> bf16 pairwise conversion (emits v_cvt_pk_bf16_f32, 4B stores)
__global__ __launch_bounds__(256) void cvt2_k(const float* __restrict__ s,
                                              __bf16* __restrict__ d, int pairs) {
  for (int i = blockIdx.x * 256 + threadIdx.x; i < pairs; i += gridDim.x * 256) {
    float2 f = ((const float2*)s)[i];
    bvec2 o;
    o[0] = (__bf16)f.x;
    o[1] = (__bf16)f.y;
    ((bvec2*)d)[i] = o;
  }
}

// ---------------------------------------------------------------------------
// Pure-bf16 WMMA GEMM: C = act(A[M,K] * op(B) + bias), f32 accumulate.
//   BT=true : B stored [N,K] row-major -> one 32B contiguous vector per frag
//   BT=false: B stored [K,N] row-major -> strided b16 loads (probs x V only)
//   OutT = float or __bf16 (epilogue store type)
// WG: 256 thr (8 waves), tile 128x64, each wave 32x32 (2x2 WMMA / k-step).
// Batched over blockIdx.z (zb = z/nh, zh = z%nh). Dims are exact multiples.
// ---------------------------------------------------------------------------
template <bool BT, int EPI, typename OutT>   // EPI: 0 none, 1 +bias, 2 +bias+relu
__global__ __launch_bounds__(256) void gemm_bf16_k(
    const __bf16* __restrict__ A, long long aOffB, long long aOffH, int lda,
    const __bf16* __restrict__ Bm, long long bOffB, long long bOffH, int ldb,
    OutT* __restrict__ C, long long cOffB, long long cOffH, int ldc,
    const float* __restrict__ bias, int K, int nh) {
  const int tid  = threadIdx.x;
  const int wave = tid >> 5;
  const int lane = tid & 31;
  const int hi   = lane >> 4;
  const int ln   = lane & 15;

  const int zb = blockIdx.z / nh;
  const int zh = blockIdx.z % nh;
  A  += (long long)zb * aOffB + (long long)zh * aOffH;
  Bm += (long long)zb * bOffB + (long long)zh * bOffH;
  C  += (long long)zb * cOffB + (long long)zh * cOffH;

  const int m_base = blockIdx.x * 128 + (wave >> 1) * 32;
  const int n_base = blockIdx.y * 64  + (wave & 1)  * 32;

  v8f acc[2][2] = {};

  for (int kk = 0; kk < K; kk += 32) {
    // A fragment: lane holds row m; K elems kk + hi*16 + e
    bvec16 af[2];
#pragma unroll
    for (int i = 0; i < 2; ++i) {
      const __bf16* ap = A + (size_t)(m_base + i * 16 + ln) * lda + kk + hi * 16;
      bvec8 lo = *(const bvec8*)ap;
      bvec8 hh = *(const bvec8*)(ap + 8);
      af[i] = __builtin_shufflevector(lo, hh, 0, 1, 2, 3, 4, 5, 6, 7, 8, 9, 10,
                                      11, 12, 13, 14, 15);
    }
    // B fragment: lane holds col n; K elems kk + hi*16 + e
    bvec16 bf[2];
#pragma unroll
    for (int j = 0; j < 2; ++j) {
      if (BT) {
        bf[j] = *(const bvec16*)(Bm + (size_t)(n_base + j * 16 + ln) * ldb + kk +
                                 hi * 16);
      } else {
        const __bf16* bp = Bm + (size_t)(kk + hi * 16) * ldb + (n_base + j * 16 + ln);
#pragma unroll
        for (int e = 0; e < 16; ++e) bf[j][e] = bp[(size_t)e * ldb];
      }
    }
#pragma unroll
    for (int i = 0; i < 2; ++i)
#pragma unroll
      for (int j = 0; j < 2; ++j) acc[i][j] = wmma_bf16(af[i], bf[j], acc[i][j]);
  }

  // Epilogue. D layout: VGPR r -> row r + 8*hi, col = lane&15.
#pragma unroll
  for (int i = 0; i < 2; ++i) {
#pragma unroll
    for (int j = 0; j < 2; ++j) {
      const int n  = n_base + j * 16 + ln;
      const float bv = (EPI >= 1) ? bias[n] : 0.0f;
#pragma unroll
      for (int r = 0; r < 8; ++r) {
        const int m = m_base + i * 16 + r + 8 * hi;
        float v = acc[i][j][r] + bv;
        if (EPI == 2) v = fmaxf(v, 0.0f);
        C[(size_t)m * ldc + n] = (OutT)v;
      }
    }
  }
}

// ---------------------------------------------------------------------------
// Embedding + sinusoidal positional encoding -> f32 stream + bf16 copy
// ---------------------------------------------------------------------------
__global__ __launch_bounds__(256) void embed_pe_k(const float* __restrict__ emb,
                                                  const int* __restrict__ tok,
                                                  float* __restrict__ out,
                                                  __bf16* __restrict__ outb) {
  const int row = blockIdx.x;  // b*L + l
  const int l   = row % LSEQ;
  const int t   = tok[row];
  const float pos = (float)l;
#pragma unroll
  for (int c = 0; c < 2; ++c) {
    const int d = threadIdx.x + c * 256;
    const float phase =
        pos * __expf(-(float)(d >> 1) * (9.2103403719761836f / (float)DMC)) +
        1.5707963267948966f * (float)(d & 1);
    const float v = emb[(size_t)t * DMC + d] + __sinf(phase);
    out[(size_t)row * DMC + d]  = v;
    outb[(size_t)row * DMC + d] = (__bf16)v;
  }
}

// ---------------------------------------------------------------------------
// Masked softmax over last dim (512): reads f32 scores, writes bf16 probs.
// Reference: s -> where(mask,-inf,s) -> softmax(s/sqrt(DK))
// ---------------------------------------------------------------------------
__global__ __launch_bounds__(256) void softmax_mask_k(
    const float* __restrict__ S, __bf16* __restrict__ Pr,
    const unsigned char* __restrict__ km, int causal, float scale) {
  const int row = blockIdx.x;  // (b*NH + h)*L + q
  const int q   = row % LSEQ;
  const int b   = row / (LSEQ * NHC);
  const float* rp = S + (size_t)row * LSEQ;
  __bf16* wp = Pr + (size_t)row * LSEQ;

  __shared__ float red[256];
  float x[2];
#pragma unroll
  for (int c = 0; c < 2; ++c) {
    const int k = threadIdx.x + c * 256;
    bool masked = (causal && (k > q)) || (km[b * LSEQ + k] != 0);
    x[c] = masked ? -1.0e30f : rp[k] * scale;
  }
  float mx = fmaxf(x[0], x[1]);
  red[threadIdx.x] = mx;
  __syncthreads();
  for (int s = 128; s > 0; s >>= 1) {
    if (threadIdx.x < s) red[threadIdx.x] = fmaxf(red[threadIdx.x], red[threadIdx.x + s]);
    __syncthreads();
  }
  mx = red[0];
  __syncthreads();
  const float e0 = __expf(x[0] - mx), e1 = __expf(x[1] - mx);
  red[threadIdx.x] = e0 + e1;
  __syncthreads();
  for (int s = 128; s > 0; s >>= 1) {
    if (threadIdx.x < s) red[threadIdx.x] += red[threadIdx.x + s];
    __syncthreads();
  }
  const float inv = 1.0f / red[0];
  wp[threadIdx.x]       = (__bf16)(e0 * inv);
  wp[threadIdx.x + 256] = (__bf16)(e1 * inv);
}

// ---------------------------------------------------------------------------
// O = LayerNorm(X + R)*g + b, dual-emit f32 (residual) + bf16 (next GEMM A)
// ---------------------------------------------------------------------------
__global__ __launch_bounds__(256) void add_ln_k(const float* __restrict__ X,
                                                const float* __restrict__ R,
                                                const float* __restrict__ g,
                                                const float* __restrict__ be,
                                                float* __restrict__ O,
                                                __bf16* __restrict__ Ob) {
  const int row = blockIdx.x;
  const size_t base = (size_t)row * DMC;
  __shared__ float red[256];
  float v0 = X[base + threadIdx.x]       + R[base + threadIdx.x];
  float v1 = X[base + threadIdx.x + 256] + R[base + threadIdx.x + 256];
  red[threadIdx.x] = v0 + v1;
  __syncthreads();
  for (int s = 128; s > 0; s >>= 1) {
    if (threadIdx.x < s) red[threadIdx.x] += red[threadIdx.x + s];
    __syncthreads();
  }
  const float mean = red[0] * (1.0f / (float)DMC);
  __syncthreads();
  red[threadIdx.x] = v0 * v0 + v1 * v1;
  __syncthreads();
  for (int s = 128; s > 0; s >>= 1) {
    if (threadIdx.x < s) red[threadIdx.x] += red[threadIdx.x + s];
    __syncthreads();
  }
  const float var = red[0] * (1.0f / (float)DMC) - mean * mean;
  const float rs  = rsqrtf(var + 1e-5f);
  const float o0 = (v0 - mean) * rs * g[threadIdx.x] + be[threadIdx.x];
  const float o1 = (v1 - mean) * rs * g[threadIdx.x + 256] + be[threadIdx.x + 256];
  O[base + threadIdx.x]        = o0;
  O[base + threadIdx.x + 256]  = o1;
  Ob[base + threadIdx.x]       = (__bf16)o0;
  Ob[base + threadIdx.x + 256] = (__bf16)o1;
}

// ---------------------------------------------------------------------------
// Fused logits GEMM + online softmax + NLL. Pure bf16 operands.
// WG handles 16 rows; A fragments for all K=512 hoisted (16 x v16bf); waves
// stream vocab in 128-col chunks; per-lane flash (max,sumexp); target logit
// captured by unique lane; shuffle + LDS combine; atomic accumulate.
// ---------------------------------------------------------------------------
__global__ __launch_bounds__(256) void logits_loss_k(
    const __bf16* __restrict__ A,    // demb bf16 [4096,512]
    const __bf16* __restrict__ W,    // out_W bf16 [32000,512]
    const float* __restrict__ bias,  // out_b f32 [32000]
    const int* __restrict__ target, const unsigned char* __restrict__ tmask,
    float* __restrict__ accum) {
  const int tid  = threadIdx.x;
  const int wave = tid >> 5;
  const int lane = tid & 31;
  const int hi   = lane >> 4;
  const int ln   = lane & 15;
  const int m0   = blockIdx.x * 16;

  __shared__ float  s_t[16];
  __shared__ float2 s_ms[8][16];
  if (tid < 16) s_t[tid] = 0.0f;
  __syncthreads();

  bvec16 af[16];
#pragma unroll
  for (int ks = 0; ks < 16; ++ks) {
    const __bf16* ap = A + (size_t)(m0 + ln) * DMC + ks * 32 + hi * 16;
    bvec8 lo = *(const bvec8*)ap;
    bvec8 hh = *(const bvec8*)(ap + 8);
    af[ks] = __builtin_shufflevector(lo, hh, 0, 1, 2, 3, 4, 5, 6, 7, 8, 9, 10,
                                     11, 12, 13, 14, 15);
  }

  int tcol[8];
#pragma unroll
  for (int r = 0; r < 8; ++r) {
    const int R = m0 + r + 8 * hi;
    const int l = R & (LSEQ - 1);
    tcol[r] = (l < LSEQ - 1) ? target[(R >> 9) * LSEQ + l + 1] : -1;
  }

  float runM[8], runS[8], tl[8];
  bool  cap[8];
#pragma unroll
  for (int r = 0; r < 8; ++r) { runM[r] = -3.0e38f; runS[r] = 0.0f; tl[r] = 0.0f; cap[r] = false; }

  for (int n0 = wave * 16; n0 < VOCAB; n0 += 128) {
    v8f acc = {};
#pragma unroll
    for (int ks = 0; ks < 16; ++ks) {
      bvec16 bf = *(const bvec16*)(W + (size_t)(n0 + ln) * DMC + ks * 32 + hi * 16);
      acc = wmma_bf16(af[ks], bf, acc);
    }
    const int   n  = n0 + ln;
    const float bv = bias[n];
#pragma unroll
    for (int r = 0; r < 8; ++r) {
      const float lg = acc[r] + bv;
      if (n == tcol[r]) { tl[r] = lg; cap[r] = true; }
      if (lg > runM[r]) {
        runS[r] = runS[r] * __expf(runM[r] - lg) + 1.0f;
        runM[r] = lg;
      } else {
        runS[r] += __expf(lg - runM[r]);
      }
    }
  }

#pragma unroll
  for (int r = 0; r < 8; ++r)
    if (cap[r]) s_t[r + 8 * hi] = tl[r];

#pragma unroll
  for (int r = 0; r < 8; ++r) {
    float m = runM[r], s = runS[r];
#pragma unroll
    for (int off = 1; off < 16; off <<= 1) {
      float m2 = __shfl_xor(m, off);
      float s2 = __shfl_xor(s, off);
      float mx = fmaxf(m, m2);
      s = s * __expf(m - mx) + s2 * __expf(m2 - mx);
      m = mx;
    }
    if (ln == 0) s_ms[wave][hi * 8 + r] = make_float2(m, s);
  }
  __syncthreads();

  if (tid < 16) {
    float m = s_ms[0][tid].x, s = s_ms[0][tid].y;
#pragma unroll
    for (int w = 1; w < 8; ++w) {
      float m2 = s_ms[w][tid].x, s2 = s_ms[w][tid].y;
      float mx = fmaxf(m, m2);
      s = s * __expf(m - mx) + s2 * __expf(m2 - mx);
      m = mx;
    }
    const int R = m0 + tid;
    const int l = R & (LSEQ - 1);
    if (l < LSEQ - 1 && tmask[(R >> 9) * LSEQ + l + 1] == 0) {
      const float p  = __expf(s_t[tid] - m) / s;
      atomicAdd(&accum[0], -logf(p + 1e-7f));
      atomicAdd(&accum[1], 1.0f);
    }
  }
}

__global__ void init_accum_k(float* accum) { accum[0] = 0.0f; accum[1] = 0.0f; }
__global__ void finalize_k(const float* accum, float* out) { out[0] = accum[0] / accum[1]; }

// ---------------------------------------------------------------------------
// Host orchestration
// ---------------------------------------------------------------------------
extern "C" void kernel_launch(void* const* d_in, const int* in_sizes, int n_in,
                              void* d_out, int out_size, void* d_ws, size_t ws_size,
                              hipStream_t stream) {
  (void)n_in; (void)out_size; (void)ws_size;

  auto P = [&](int i) { return (const float*)d_in[i]; };
  // JAX pytree (sorted dict keys) leaf ordering:
  // dec_blocks[j] (base 18j): W1,W2,b1,b2, cross{Wk,Wo,Wq,Wv}, ln1_b,ln1_g,
  //   ln2_b,ln2_g,ln3_b,ln3_g, self{Wk,Wo,Wq,Wv}
  // dec_emb=108; enc_blocks[j] (base 109+12j): W1,W2,b1,b2,ln1_b,ln1_g,ln2_b,
  //   ln2_g, mha{Wk,Wo,Wq,Wv}
  // enc_emb=181; out_W=182; out_b=183; source=184; src_mask=185;
  // tar_mask=186; target=187
  const int* src_tok = (const int*)d_in[184];
  const int* tgt_tok = (const int*)d_in[187];
  const unsigned char* src_mask = (const unsigned char*)d_in[185];
  const unsigned char* tar_mask = (const unsigned char*)d_in[186];

  // ---------------- workspace layout ----------------
  float* ws = (float*)d_ws;
  const size_t SZ_ACT = (size_t)BL * DMC;                  // 2,097,152
  const size_t SZ_SC  = (size_t)BATCH * NHC * LSEQ * LSEQ; // 16,777,216
  float* x_enc  = ws;                 // f32 residual streams + scores
  float* x_dec  = x_enc + SZ_ACT;
  float* tmp    = x_dec + SZ_ACT;
  float* scores = tmp   + SZ_ACT;
  float* accum  = scores + SZ_SC;

  __bf16* bfb    = (__bf16*)(accum + 16);  // bf16 region (aligned: all sizes big pow2-ish)
  __bf16* xe_bf  = bfb;                 xe_bf  += 0;
  __bf16* xd_bf  = xe_bf  + SZ_ACT;
  __bf16* Qh_bf  = xd_bf  + SZ_ACT;
  __bf16* Kh_bf  = Qh_bf  + SZ_ACT;
  __bf16* Vh_bf  = Kh_bf  + SZ_ACT;
  __bf16* hc_bf  = Vh_bf  + SZ_ACT;
  __bf16* ffh_bf = hc_bf  + SZ_ACT;                  // [BL, DFF]
  __bf16* pr_bf  = ffh_bf + (size_t)BL * DFFC;       // probs [B,NH,L,L]
  __bf16* warena = pr_bf  + SZ_SC;                   // bf16 weight arena

  // ---------------- weight bf16 conversion (once per call) ----------------
  long long wofs[188];
  for (int i = 0; i < 188; ++i) wofs[i] = -1;
  int mats[104]; int nm = 0;
  for (int j = 0; j < NBLK; ++j) {
    const int b0 = j * 18;
    const int ids[10] = {b0+0, b0+1, b0+4, b0+5, b0+6, b0+7, b0+14, b0+15, b0+16, b0+17};
    for (int t = 0; t < 10; ++t) mats[nm++] = ids[t];
  }
  for (int j = 0; j < NBLK; ++j) {
    const int b0 = 109 + j * 12;
    const int ids[6] = {b0+0, b0+1, b0+8, b0+9, b0+10, b0+11};
    for (int t = 0; t < 6; ++t) mats[nm++] = ids[t];
  }
  mats[nm++] = 182;  // out_W
  long long cursor = 0;
  for (int t = 0; t < nm; ++t) { wofs[mats[t]] = cursor; cursor += in_sizes[mats[t]]; }
  for (int t = 0; t < nm; ++t) {
    const int idx = mats[t];
    const int pairs = in_sizes[idx] / 2;
    int grid = (pairs + 255) / 256; if (grid > 2048) grid = 2048;
    cvt2_k<<<grid, 256, 0, stream>>>(P(idx), warena + wofs[idx], pairs);
  }
  auto WB = [&](int idx) { return (const __bf16*)(warena + wofs[idx]); };

  const long long ZA  = (long long)LSEQ * DMC;   // per-b activation offset
  const long long ZS  = (long long)LSEQ * LSEQ;  // per-h score offset
  const long long ZSB = (long long)NHC * ZS;     // per-b score offset

  // ---------------- GEMM wrappers ----------------
  dim3 blk(256);
  // x[BL,512] * W^T -> bf16 [BL,512]
  auto G_proj = [&](const __bf16* A, const __bf16* W, __bf16* C) {
    gemm_bf16_k<true, 0, __bf16><<<dim3(BL/128, DMC/64, 1), blk, 0, stream>>>(
        A, 0, 0, DMC, W, 0, 0, DMC, C, 0, 0, DMC, nullptr, DMC, 1);
  };

  auto mha = [&](const __bf16* xq, const __bf16* xkv, const __bf16* Wq,
                 const __bf16* Wk, const __bf16* Wv, const __bf16* Wo,
                 int causal, const unsigned char* kmask) {
    G_proj(xq,  Wq, Qh_bf);
    G_proj(xkv, Wk, Kh_bf);
    G_proj(xkv, Wv, Vh_bf);
    // scores[b,h] = Qh[b,:,h*64:] * Kh[b,:,h*64:]^T  (M=N=512, K=64) -> f32
    gemm_bf16_k<true, 0, float><<<dim3(LSEQ/128, LSEQ/64, BATCH*NHC), blk, 0, stream>>>(
        Qh_bf, ZA, 64, DMC, Kh_bf, ZA, 64, DMC,
        scores, ZSB, ZS, LSEQ, nullptr, 64, NHC);
    softmax_mask_k<<<BATCH * NHC * LSEQ, blk, 0, stream>>>(scores, pr_bf, kmask,
                                                           causal, 0.125f);
    // hcat[b,:,h*64:] = probs[b,h] * Vh[b,:,h*64:]   (M=512,N=64,K=512) -> bf16
    gemm_bf16_k<false, 0, __bf16><<<dim3(LSEQ/128, 1, BATCH*NHC), blk, 0, stream>>>(
        pr_bf, ZSB, ZS, LSEQ, Vh_bf, ZA, 64, DMC,
        hc_bf, ZA, 64, DMC, nullptr, LSEQ, NHC);
    // tmp = hcat * Wo^T -> f32
    gemm_bf16_k<true, 0, float><<<dim3(BL/128, DMC/64, 1), blk, 0, stream>>>(
        hc_bf, 0, 0, DMC, Wo, 0, 0, DMC, tmp, 0, 0, DMC, nullptr, DMC, 1);
  };

  auto ffn = [&](const __bf16* xb, const float* W1o, const float* W2o,
                 const __bf16* W1, const __bf16* W2, const float* b1,
                 const float* b2) {
    (void)W1o; (void)W2o;
    gemm_bf16_k<true, 2, __bf16><<<dim3(BL/128, DFFC/64, 1), blk, 0, stream>>>(
        xb, 0, 0, DMC, W1, 0, 0, DMC, ffh_bf, 0, 0, DFFC, b1, DMC, 1);
    gemm_bf16_k<true, 1, float><<<dim3(BL/128, DMC/64, 1), blk, 0, stream>>>(
        ffh_bf, 0, 0, DFFC, W2, 0, 0, DFFC, tmp, 0, 0, DMC, b2, DFFC, 1);
  };

  init_accum_k<<<1, 1, 0, stream>>>(accum);

  embed_pe_k<<<BL, blk, 0, stream>>>(P(181), src_tok, x_enc, xe_bf);  // enc_emb
  embed_pe_k<<<BL, blk, 0, stream>>>(P(108), tgt_tok, x_dec, xd_bf);  // dec_emb

  // ---- Encoder ----
  for (int j = 0; j < NBLK; ++j) {
    const int b0 = 109 + j * 12;
    mha(xe_bf, xe_bf, WB(b0+10), WB(b0+8), WB(b0+11), WB(b0+9), 0, src_mask);
    add_ln_k<<<BL, blk, 0, stream>>>(tmp, x_enc, P(b0+5), P(b0+4), x_enc, xe_bf);
    ffn(xe_bf, P(b0+0), P(b0+1), WB(b0+0), WB(b0+1), P(b0+2), P(b0+3));
    add_ln_k<<<BL, blk, 0, stream>>>(tmp, x_enc, P(b0+7), P(b0+6), x_enc, xe_bf);
  }

  // ---- Decoder ---- (x_enc/xe_bf now hold enc_out; untouched below)
  for (int j = 0; j < NBLK; ++j) {
    const int b0 = j * 18;
    mha(xd_bf, xd_bf, WB(b0+16), WB(b0+14), WB(b0+17), WB(b0+15), 1, tar_mask);
    add_ln_k<<<BL, blk, 0, stream>>>(tmp, x_dec, P(b0+9), P(b0+8), x_dec, xd_bf);
    mha(xd_bf, xe_bf, WB(b0+6), WB(b0+4), WB(b0+7), WB(b0+5), 0, src_mask);
    add_ln_k<<<BL, blk, 0, stream>>>(tmp, x_dec, P(b0+11), P(b0+10), x_dec, xd_bf);
    ffn(xd_bf, P(b0+0), P(b0+1), WB(b0+0), WB(b0+1), P(b0+2), P(b0+3));
    add_ln_k<<<BL, blk, 0, stream>>>(tmp, x_dec, P(b0+13), P(b0+12), x_dec, xd_bf);
  }

  // ---- Fused logits + softmax + NLL loss ----
  logits_loss_k<<<BL / 16, blk, 0, stream>>>(xd_bf, WB(182), P(183), tgt_tok,
                                             tar_mask, accum);
  finalize_k<<<1, 1, 0, stream>>>(accum, (float*)d_out);
}
